// ConvNet_81097572483331
// MI455X (gfx1250) — compile-verified, matrix operand order checked by reading, not run on hardware
//
#include <hip/hip_runtime.h>
#include <hip/hip_bf16.h>
#include <math.h>

typedef __attribute__((ext_vector_type(16))) _Float16 v16h;
typedef __attribute__((ext_vector_type(8)))  _Float16 v8h;
typedef __attribute__((ext_vector_type(8)))  float    v8f;
typedef int v4i __attribute__((vector_size(16)));

#define NBATCH 2048
#define SIGLEN 368
#define NFILT  112               // valid filters (7 x 16)
#define MROWS  128               // padded to 8 x 16 so all 8 waves run WMMA
#define NSEL   112
#define NCOLS  (NBATCH * NSEL)   // 229376 = 3584 x 64
#define COLPAD 40                // halves per LDS column slot (32 data + 8 pad = 80B)

// ---------------- async-copy / wait helpers (guarded; degrade gracefully) ----
#if defined(__has_builtin)
#if __has_builtin(__builtin_amdgcn_global_load_async_to_lds_b128)
#define HAVE_ASYNC_LDS 1
#endif
#if __has_builtin(__builtin_amdgcn_s_wait_asynccnt)
#define WAIT_ASYNC0() __builtin_amdgcn_s_wait_asynccnt(0)
#endif
#endif
#ifndef WAIT_ASYNC0
#define WAIT_ASYNC0() asm volatile("s_wait_asynccnt 0x0" ::: "memory")
#endif

// flat -> AS(1)/AS(3) v4i* casts for the async builtin.
// (LDS offset == low 32 bits of the flat address; global flat == global addr.)
__device__ __forceinline__ __attribute__((address_space(1))) v4i*
to_glb_v4i(const void* p) {
  return (__attribute__((address_space(1))) v4i*)(unsigned long long)p;
}
__device__ __forceinline__ __attribute__((address_space(3))) v4i*
to_lds_v4i(void* p) {
  return (__attribute__((address_space(3))) v4i*)(unsigned int)(unsigned long long)p;
}

// ---------------------------------------------------------------------------
// Pack wavelet bank (112 x K f32) -> Ah (128 x Kpad f16, zero tail rows/cols)
// ---------------------------------------------------------------------------
__global__ void pack_wavelets(const float* __restrict__ W, _Float16* __restrict__ Ah,
                              int K, int Kpad) {
  int tid = blockIdx.x * blockDim.x + threadIdx.x;
  int total = MROWS * Kpad;
  if (tid >= total) return;
  int f = tid / Kpad;
  int k = tid - f * Kpad;
  float v = (f < NFILT && k < K) ? W[(long)f * K + k] : 0.f;
  Ah[tid] = (_Float16)v;
}

// ---------------------------------------------------------------------------
// im2col: Xp[n][k] = x[b][sel(j)+k-pad] (SAME padding), n = b*112 + j
// IMG_SELECT[j] == floor(j*367/111).
// ---------------------------------------------------------------------------
__global__ void pack_patches(const float* __restrict__ x, _Float16* __restrict__ Xp,
                             int K, int Kpad) {
  int tid = blockIdx.x * blockDim.x + threadIdx.x;
  int k8cnt = Kpad >> 3;
  long total = (long)NCOLS * k8cnt;
  if (tid >= total) return;
  int n  = tid / k8cnt;
  int k0 = (tid - n * k8cnt) << 3;
  int b = n / NSEL, j = n - b * NSEL;
  int sel = (j * 367) / 111;
  int pad = (K - 1) >> 1;
  const float* xr = x + (long)b * SIGLEN;
  v8h out;
#pragma unroll
  for (int i = 0; i < 8; ++i) {
    int k = k0 + i;
    int src = sel + k - pad;
    float v = (k < K && src >= 0 && src < SIGLEN) ? xr[src] : 0.f;
    out[i] = (_Float16)v;
  }
  *reinterpret_cast<v8h*>(Xp + (long)n * Kpad + k0) = out;
}

// ---------------------------------------------------------------------------
// Wavelet GEMM: C[112, 229376] = Ah x Xp^T with v_wmma_f32_16x16x32_f16.
// Block = 256 threads = 8 waves; block owns a 64-column group. Every wave
// computes one 16-row M-tile (wave 7's tile is the zero-padded rows 112-127,
// so the hot loop has NO divergence and EXEC stays all-ones for WMMA).
// B chunks (64 cols x 32 K halves) double-buffer in LDS via async copies:
// 256 threads x 16B = 4KB = one chunk, tracked by ASYNCcnt.
// ---------------------------------------------------------------------------
__global__ void __launch_bounds__(256)
wavelet_gemm_wmma(const _Float16* __restrict__ Ah, const _Float16* __restrict__ Xp,
                  float* __restrict__ img, int Kpad) {
  __shared__ _Float16 bbuf[2][64 * COLPAD];   // 2 x 5120 B

  const int tid  = threadIdx.x;
  const int wave = tid >> 5;
  const int lane = tid & 31;
  const int half = lane >> 4;          // K sub-offset selector (0 or 8 halves)
  const int r    = lane & 15;
  const int ncol0 = blockIdx.x * 64;

  // staging role: thread -> (column, 16B segment)
  const int sc   = tid >> 2;           // 0..63
  const int sseg = tid & 3;            // 0..3
  const _Float16* gsrc = Xp + (long)(ncol0 + sc) * Kpad + sseg * 8;
  _Float16* lds0 = &bbuf[0][sc * COLPAD + sseg * 8];
  _Float16* lds1 = &bbuf[1][sc * COLPAD + sseg * 8];

  auto stage = [&](int buf, int k0) {
    _Float16* dst = buf ? lds1 : lds0;
#ifdef HAVE_ASYNC_LDS
    __builtin_amdgcn_global_load_async_to_lds_b128(
        to_glb_v4i(gsrc + k0), to_lds_v4i(dst), 0, 0);
#else
    *reinterpret_cast<v8h*>(dst) = *reinterpret_cast<const v8h*>(gsrc + k0);
#endif
  };

  // A row for this wave's M-tile (wave 7 -> zero rows 112-127)
  const _Float16* aRow = Ah + (long)(wave * 16 + r) * Kpad + 8 * half;

  v8f acc[4] = {v8f{}, v8f{}, v8f{}, v8f{}};
  const int nch = Kpad >> 5;

  union Frag { v16h v; v8h h[2]; };

  stage(0, 0);                         // prologue: chunk 0 -> buf 0
  for (int i = 0; i < nch; ++i) {
    const int cur = i & 1;
    WAIT_ASYNC0();                     // my chunk-i transfers done
    __syncthreads();                   // everyone's chunk-i transfers done
    if (i + 1 < nch) stage(cur ^ 1, (i + 1) << 5);  // overlap next chunk

    // Issue all fragment loads first so waits can count down instead of 0.
    Frag a;
    const v8h* ap = reinterpret_cast<const v8h*>(aRow + ((long)i << 5));
    a.h[0] = ap[0];                    // K = 8h   .. 8h+7
    a.h[1] = ap[2];                    // K = 16+8h.. 16+8h+7
    Frag bf[4];
#pragma unroll
    for (int t = 0; t < 4; ++t) {
      const _Float16* cb = &bbuf[cur][(t * 16 + r) * COLPAD + 8 * half];
      bf[t].h[0] = *reinterpret_cast<const v8h*>(cb);
      bf[t].h[1] = *reinterpret_cast<const v8h*>(cb + 16);
    }
#pragma unroll
    for (int t = 0; t < 4; ++t)
      acc[t] = __builtin_amdgcn_wmma_f32_16x16x32_f16(
          false, a.v, false, bf[t].v, (short)0, acc[t], false, false);
  }

  if (wave < 7) {                      // wave 7 computed zeros; discard
    // D layout: VGPR v -> M = v + 8*half, N = r
    int fbase = wave * 16 + 8 * half;
#pragma unroll
    for (int t = 0; t < 4; ++t) {
      int n = ncol0 + t * 16 + r;
      int b = n / NSEL, j = n - b * NSEL;
      float* orow = img + (long)b * NFILT * NSEL + j;
#pragma unroll
      for (int v = 0; v < 8; ++v)
        orow[(long)(fbase + v) * NSEL] = acc[t][v];
    }
  }
}

// ---------------------------------------------------------------------------
// Fused 3x3 SAME conv + bias + ReLU + 2x2/2 maxpool, templated for full unroll.
// One thread per pooled pixel computes ALL COUT channels: each 4x4 input
// window is loaded once per ci and reused for COUT*4 positions (MAC:load ~27).
// ---------------------------------------------------------------------------
template <int CIN, int COUT, int H, int W>
__global__ void conv3x3_relu_pool(const float* __restrict__ in, const float* __restrict__ w,
                                  const float* __restrict__ bias, float* __restrict__ out) {
  constexpr int OH = H / 2, OW = W / 2;
  const long total = (long)NBATCH * OH * OW;
  long tid = (long)blockIdx.x * blockDim.x + threadIdx.x;
  if (tid >= total) return;
  int ox = (int)(tid % OW); long t1 = tid / OW;
  int oy = (int)(t1 % OH);
  int b  = (int)(t1 / OH);
  int y0 = 2 * oy, x0 = 2 * ox;

  float acc[COUT][4];
#pragma unroll
  for (int co = 0; co < COUT; ++co) {
    float bv = bias[co];
#pragma unroll
    for (int p = 0; p < 4; ++p) acc[co][p] = bv;
  }

  const float* inb = in + (long)b * CIN * H * W;
#pragma unroll
  for (int ci = 0; ci < CIN; ++ci) {
    const float* ip = inb + (long)ci * H * W;
    float pix[4][4];
#pragma unroll
    for (int yy = 0; yy < 4; ++yy) {
      int iy = y0 + yy - 1;
#pragma unroll
      for (int xx = 0; xx < 4; ++xx) {
        int ix = x0 + xx - 1;
        pix[yy][xx] = (iy >= 0 && iy < H && ix >= 0 && ix < W) ? ip[(long)iy * W + ix] : 0.f;
      }
    }
#pragma unroll
    for (int co = 0; co < COUT; ++co) {
      const float* wp = w + ((long)co * CIN + ci) * 9;
      float wr[9];
#pragma unroll
      for (int q = 0; q < 9; ++q) wr[q] = wp[q];
#pragma unroll
      for (int dy = 0; dy < 2; ++dy)
#pragma unroll
      for (int dx = 0; dx < 2; ++dx)
#pragma unroll
      for (int ky = 0; ky < 3; ++ky)
#pragma unroll
      for (int kx = 0; kx < 3; ++kx)
        acc[co][dy * 2 + dx] =
            fmaf(pix[dy + ky][dx + kx], wr[ky * 3 + kx], acc[co][dy * 2 + dx]);
    }
  }

  float* ob = out + (long)b * COUT * OH * OW + (long)oy * OW + ox;
#pragma unroll
  for (int co = 0; co < COUT; ++co) {
    float m = acc[co][0];
#pragma unroll
    for (int p = 1; p < 4; ++p) m = acc[co][p] > m ? acc[co][p] : m;
    m = m > 0.f ? m : 0.f;           // relu-then-max == max-then-relu
    ob[(long)co * OH * OW] = m;
  }
}

// ---------------------------------------------------------------------------
__global__ void fc1_relu(const float* __restrict__ in, const float* __restrict__ w,
                         const float* __restrict__ bias, float* __restrict__ out) {
  int tid = blockIdx.x * blockDim.x + threadIdx.x;
  if (tid >= NBATCH * 48) return;
  int o = tid % 48, b = tid / 48;
  const float* iv = in + (long)b * 588;
  const float* wv = w + (long)o * 588;
  float s = bias[o];
  for (int i = 0; i < 588; ++i) s = fmaf(iv[i], wv[i], s);
  out[tid] = s > 0.f ? s : 0.f;
}

__global__ void fc2_softmax(const float* __restrict__ in, const float* __restrict__ w,
                            const float* __restrict__ bias, float* __restrict__ out) {
  int b = blockIdx.x * blockDim.x + threadIdx.x;
  if (b >= NBATCH) return;
  const float* iv = in + (long)b * 48;
  float z[4];
  float mx = -INFINITY;
#pragma unroll
  for (int o = 0; o < 4; ++o) {
    float s = bias[o];
    const float* wv = w + o * 48;
    for (int i = 0; i < 48; ++i) s = fmaf(iv[i], wv[i], s);
    z[o] = s;
    mx = s > mx ? s : mx;
  }
  float den = 0.f;
#pragma unroll
  for (int o = 0; o < 4; ++o) { z[o] = expf(z[o] - mx); den += z[o]; }
  float inv = 1.f / den;
#pragma unroll
  for (int o = 0; o < 4; ++o) out[(long)b * 4 + o] = z[o] * inv;
}

// ---------------------------------------------------------------------------
extern "C" void kernel_launch(void* const* d_in, const int* in_sizes, int n_in,
                              void* d_out, int out_size, void* d_ws, size_t ws_size,
                              hipStream_t stream) {
  const float* x    = (const float*)d_in[0];
  const float* wav  = (const float*)d_in[1];
  const float* c1w  = (const float*)d_in[2];  const float* c1b = (const float*)d_in[3];
  const float* c2w  = (const float*)d_in[4];  const float* c2b = (const float*)d_in[5];
  const float* c3w  = (const float*)d_in[6];  const float* c3b = (const float*)d_in[7];
  const float* c4w  = (const float*)d_in[8];  const float* c4b = (const float*)d_in[9];
  const float* f1w  = (const float*)d_in[10]; const float* f1b = (const float*)d_in[11];
  const float* f2w  = (const float*)d_in[12]; const float* f2b = (const float*)d_in[13];

  int K    = in_sizes[1] / NFILT;       // wavelet filter length (~559, odd)
  int Kpad = (K + 31) & ~31;            // multiple of 32 for WMMA K-chunks

  char* ws = (char*)d_ws;
  size_t off = 0;
  auto alloc = [&](size_t bytes) -> size_t {
    size_t o = off; off = (off + bytes + 255) & ~(size_t)255; return o;
  };
  size_t ah_off  = alloc((size_t)MROWS * Kpad * 2);
  size_t img_off = alloc((size_t)NBATCH * NFILT * NSEL * 4);   // ~103 MB
  size_t xp_off  = alloc((size_t)NCOLS * Kpad * 2);            // ~264 MB (tail)

  // Conv intermediate chain reuses the Xp region (dead after the GEMM).
  size_t hoff = xp_off;
  auto halloc = [&](size_t bytes) -> size_t {
    size_t o = hoff; hoff = (hoff + bytes + 255) & ~(size_t)255; return o;
  };
  size_t h1_off = halloc((size_t)NBATCH *  6 * 56 * 56 * 4);
  size_t h2_off = halloc((size_t)NBATCH * 12 * 28 * 28 * 4);
  size_t h3_off = halloc((size_t)NBATCH * 12 * 14 * 14 * 4);
  size_t h4_off = halloc((size_t)NBATCH * 12 *  7 *  7 * 4);
  size_t g1_off = halloc((size_t)NBATCH * 48 * 4);

  _Float16* Ah  = (_Float16*)(ws + ah_off);
  _Float16* Xp  = (_Float16*)(ws + xp_off);
  float*    img = (float*)(ws + img_off);
  float*    h1  = (float*)(ws + h1_off);
  float*    h2  = (float*)(ws + h2_off);
  float*    h3  = (float*)(ws + h3_off);
  float*    h4  = (float*)(ws + h4_off);
  float*    g1  = (float*)(ws + g1_off);

  const int tpb = 256;
  long n;

  n = (long)MROWS * Kpad;
  pack_wavelets<<<(int)((n + tpb - 1) / tpb), tpb, 0, stream>>>(wav, Ah, K, Kpad);

  n = (long)NCOLS * (Kpad >> 3);
  pack_patches<<<(int)((n + tpb - 1) / tpb), tpb, 0, stream>>>(x, Xp, K, Kpad);

  // one block per 64-column group
  wavelet_gemm_wmma<<<NCOLS / 64, 256, 0, stream>>>(Ah, Xp, img, Kpad);

  n = (long)NBATCH * 56 * 56;
  conv3x3_relu_pool<1, 6, 112, 112>
      <<<(int)((n + tpb - 1) / tpb), tpb, 0, stream>>>(img, c1w, c1b, h1);
  n = (long)NBATCH * 28 * 28;
  conv3x3_relu_pool<6, 12, 56, 56>
      <<<(int)((n + tpb - 1) / tpb), tpb, 0, stream>>>(h1, c2w, c2b, h2);
  n = (long)NBATCH * 14 * 14;
  conv3x3_relu_pool<12, 12, 28, 28>
      <<<(int)((n + tpb - 1) / tpb), tpb, 0, stream>>>(h2, c3w, c3b, h3);
  n = (long)NBATCH * 7 * 7;
  conv3x3_relu_pool<12, 12, 14, 14>
      <<<(int)((n + tpb - 1) / tpb), tpb, 0, stream>>>(h3, c4w, c4b, h4);

  n = (long)NBATCH * 48;
  fc1_relu<<<(int)((n + tpb - 1) / tpb), tpb, 0, stream>>>(h4, f1w, f1b, g1);

  fc2_softmax<<<(NBATCH + tpb - 1) / tpb, tpb, 0, stream>>>(g1, f2w, f2b, (float*)d_out);
}
